// SequenceModel_41188736369289
// MI455X (gfx1250) — compile-verified, matrix operand order checked by reading, not run on hardware
//
#include <hip/hip_runtime.h>
#include <hip/hip_bf16.h>
#include <stdint.h>

// ---------------- problem constants ----------------
#define T_STEPS 100
#define NBATCH  128
#define SDIM    128
#define ADIM    32
#define HDIM    1024   // L == Z == 1024
#define KT_H    (HDIM / 32)   // 32 k-tiles in the recurrent reduction

typedef __bf16 bf16_t;
typedef bf16_t v16bf __attribute__((ext_vector_type(16)));
typedef float  v8f   __attribute__((ext_vector_type(8)));

__device__ __forceinline__ bf16_t f2bf(float f) {
  uint32_t u = __builtin_bit_cast(uint32_t, f);
  uint32_t r = u + 0x7FFFu + ((u >> 16) & 1u);     // round-to-nearest-even
  uint16_t h = (uint16_t)(r >> 16);
  return __builtin_bit_cast(bf16_t, h);
}

__device__ __forceinline__ float sigmoidf_(float x) {
  return 1.0f / (1.0f + __expf(-x));
}

// Map (row-in-16x32-tile r, k-in-tile kloc) -> (lane, elem) of the CDNA5
// 16-bit A-fragment layout: lane m=r (+16 if khalf), elems cover two 8-wide
// contiguous k runs: k = khalf*8 + e (e<8), k = 16 + khalf*8 + (e-8) (e>=8).
__device__ __forceinline__ void amap(int r, int kloc, int& lane, int& e) {
  int khalf = (kloc >> 3) & 1;
  lane = r + (khalf << 4);
  e = (kloc < 16) ? (kloc - (khalf << 3)) : (8 + (kloc - 16) - (khalf << 3));
}

// fp32 row -> bf16 A-fragment on the fly (only used by the tiny mu/lv GEMMs)
__device__ __forceinline__ v16bf load_a_frag_f32(const float* __restrict__ rowp,
                                                 int kbase, int khalf) {
  const float4* p0 = (const float4*)(rowp + kbase + khalf * 8);
  const float4* p1 = (const float4*)(rowp + kbase + 16 + khalf * 8);
  float4 a0 = p0[0], a1 = p0[1];
  float4 b0 = p1[0], b1 = p1[1];
  v16bf v;
  v[0] = f2bf(a0.x);  v[1] = f2bf(a0.y);  v[2]  = f2bf(a0.z);  v[3]  = f2bf(a0.w);
  v[4] = f2bf(a1.x);  v[5] = f2bf(a1.y);  v[6]  = f2bf(a1.z);  v[7]  = f2bf(a1.w);
  v[8] = f2bf(b0.x);  v[9] = f2bf(b0.y);  v[10] = f2bf(b0.z);  v[11] = f2bf(b0.w);
  v[12]= f2bf(b1.x);  v[13]= f2bf(b1.y);  v[14] = f2bf(b1.z);  v[15] = f2bf(b1.w);
  return v;
}

// fragment-permuted tile fetch: tile = 32 lanes x 16 bf16 contiguous (1KB)
__device__ __forceinline__ v16bf load_tile(const bf16_t* __restrict__ P,
                                           size_t tile, int lane) {
  return *(const v16bf*)(P + ((tile << 5) + lane) * 16);
}

__device__ __forceinline__ v8f wmma_bf16(v16bf a, v16bf b, v8f c) {
  return __builtin_amdgcn_wmma_f32_16x16x32_bf16(false, a, false, b,
                                                 (short)0, c, false, false);
}

// ---------------- split points ----------------------------------------------
__global__ void split_points_kernel(const float* __restrict__ s,
                                    int* __restrict__ endp,
                                    int* __restrict__ startp) {
  int n = threadIdx.x;
  if (n >= NBATCH) return;
  int end = 0;
  for (int t = 1; t < T_STEPS; ++t) {
    const float4* p = (const float4*)(s + ((size_t)n * T_STEPS + t) * SDIM);
    float acc = 0.0f;
    #pragma unroll 8
    for (int i = 0; i < SDIM / 4; ++i) {
      float4 v = p[i];
      acc += fabsf(v.x) + fabsf(v.y) + fabsf(v.z) + fabsf(v.w);
    }
    if (acc == 0.0f) { end = t - 1; break; }
  }
  endp[n]   = end;
  startp[n] = end + 2;
}

// ---------------- weight fp32 -> fragment-permuted bf16 ----------------------
__global__ void permute_weights_kernel(const float* __restrict__ W,
                                       bf16_t* __restrict__ Wp, int R, int K) {
  int tid = blockIdx.x * blockDim.x + threadIdx.x;
  if (tid >= R * K) return;
  int tile   = tid >> 9;
  int within = tid & 511;
  int lane   = within >> 4;
  int e      = within & 15;
  int ktiles = K >> 5;
  int jt = tile / ktiles;
  int kt = tile - jt * ktiles;
  int srow = jt * 16 + (lane & 15);
  int scol = kt * 32 + ((e >> 3) << 4) + (e & 7) + ((lane >> 4) << 3);
  Wp[tid] = f2bf(W[(size_t)srow * K + scol]);
}

// ---------------- pack encoder inputs concat(s,a) -> permuted bf16 -----------
// xP tiles: ((t*8 + mt)*5 + kt), K = 160
__global__ void pack_xenc_kernel(const float* __restrict__ s,
                                 const float* __restrict__ a,
                                 bf16_t* __restrict__ xP) {
  int idx = blockIdx.x * blockDim.x + threadIdx.x;
  if (idx >= T_STEPS * NBATCH * 160) return;
  int t   = idx / (NBATCH * 160);
  int rem = idx - t * (NBATCH * 160);
  int n   = rem / 160;
  int k   = rem - n * 160;
  float v = (k < SDIM) ? s[((size_t)n * T_STEPS + t) * SDIM + k]
                       : a[((size_t)n * T_STEPS + t) * ADIM + (k - SDIM)];
  int lane, e;
  amap(n & 15, k & 31, lane, e);
  size_t tile = (size_t)(t * 8 + (n >> 4)) * 5 + (k >> 5);
  xP[(tile << 9) + lane * 16 + e] = f2bf(v);
}

// ---------------- pack rolled decoder actions -> permuted bf16 ---------------
// aP tiles: (t*8 + mt), K = 32
__global__ void pack_adec_kernel(const float* __restrict__ a,
                                 const int* __restrict__ startp,
                                 bf16_t* __restrict__ aP) {
  int idx = blockIdx.x * blockDim.x + threadIdx.x;
  if (idx >= T_STEPS * NBATCH * ADIM) return;
  int t   = idx / (NBATCH * ADIM);
  int rem = idx - t * (NBATCH * ADIM);
  int n   = rem >> 5;
  int k   = rem & 31;
  int ts  = t + startp[n];
  if (ts >= T_STEPS) ts -= T_STEPS;                // per-sample left roll
  float v = a[((size_t)n * T_STEPS + ts) * ADIM + k];
  int lane, e;
  amap(n & 15, k, lane, e);
  aP[(((size_t)(t * 8 + (n >> 4))) << 9) + lane * 16 + e] = f2bf(v);
}

__global__ void init_h_ones_kernel(float* __restrict__ h, bf16_t* __restrict__ hP) {
  int i = blockIdx.x * blockDim.x + threadIdx.x;
  if (i < NBATCH * HDIM) {
    h[i]  = 1.0f;
    hP[i] = __builtin_bit_cast(bf16_t, (uint16_t)0x3F80);   // bf16(1.0)
  }
}

// ---------------- fused GRU step --------------------------------------------
// Block = 8 waves: 128 batch rows x 64 hidden cols; grid = HDIM/64 = 16.
// A-side fully bf16-permuted (zero conversion VALU in k-loops). Epilogue writes
// h fp32 + h permuted-bf16 (LDS-staged, coalesced b128 copy-out); decoder also
// emits dec_out in permuted bf16 for the head GEMM chain.
template <int KT_X, bool IS_ENC>
__global__ __launch_bounds__(256) void gru_step_kernel(
    const bf16_t* __restrict__ xP, const int* __restrict__ endp, int t,
    const float* __restrict__ h_in, float* __restrict__ h_out,
    const bf16_t* __restrict__ hP_in, bf16_t* __restrict__ hP_out,
    const bf16_t* __restrict__ Wih_p, const bf16_t* __restrict__ Whh_p,
    const float* __restrict__ bih, const float* __restrict__ bhh,
    float* __restrict__ hf, bf16_t* __restrict__ seqP) {
  __shared__ bf16_t stile[16][512];
  const int lane  = threadIdx.x & 31;
  const int wave  = threadIdx.x >> 5;
  const int khalf = lane >> 4;
  const int nloc  = lane & 15;
  const int mbase = wave * 16;
  const int colbase = blockIdx.x * 64;

  v8f r_acc[4], z_acc[4], xn_acc[4], hn_acc[4];
  #pragma unroll
  for (int c = 0; c < 4; ++c) {
    v8f z = {0, 0, 0, 0, 0, 0, 0, 0};
    r_acc[c] = z; z_acc[c] = z; xn_acc[c] = z; hn_acc[c] = z;
  }

  // ---- input segment: gates += x_t @ Wih^T ----
  #pragma unroll
  for (int kt = 0; kt < KT_X; ++kt) {
    v16bf af = load_tile(xP, (size_t)(t * 8 + wave) * KT_X + kt, lane);
    #pragma unroll
    for (int g = 0; g < 3; ++g) {
      #pragma unroll
      for (int c = 0; c < 4; ++c) {
        const int jt = g * (HDIM / 16) + (colbase >> 4) + c;
        v16bf bf_ = load_tile(Wih_p, (size_t)jt * KT_X + kt, lane);
        if (g == 0)      r_acc[c]  = wmma_bf16(af, bf_, r_acc[c]);
        else if (g == 1) z_acc[c]  = wmma_bf16(af, bf_, z_acc[c]);
        else             xn_acc[c] = wmma_bf16(af, bf_, xn_acc[c]);
      }
    }
  }

  // ---- recurrent segment: gates += h @ Whh^T ----
  for (int kt = 0; kt < KT_H; ++kt) {
    if (kt + 1 < KT_H)
      __builtin_prefetch(hP_in + (((size_t)(wave * KT_H + kt + 1) << 5) + lane) * 16, 0, 0);
    v16bf af = load_tile(hP_in, (size_t)wave * KT_H + kt, lane);
    #pragma unroll
    for (int g = 0; g < 3; ++g) {
      #pragma unroll
      for (int c = 0; c < 4; ++c) {
        const int jt = g * (HDIM / 16) + (colbase >> 4) + c;
        v16bf bf_ = load_tile(Whh_p, (size_t)jt * KT_H + kt, lane);
        if (g == 0)      r_acc[c]  = wmma_bf16(af, bf_, r_acc[c]);
        else if (g == 1) z_acc[c]  = wmma_bf16(af, bf_, z_acc[c]);
        else             hn_acc[c] = wmma_bf16(af, bf_, hn_acc[c]);
      }
    }
  }

  // ---- gating epilogue; C layout: VGPR v -> row v + 8*khalf, col = lane&15
  #pragma unroll
  for (int c = 0; c < 4; ++c) {
    const int col  = colbase + c * 16 + nloc;
    const float brz = bih[col]            + bhh[col];
    const float bzz = bih[HDIM + col]     + bhh[HDIM + col];
    const float bxn = bih[2 * HDIM + col];
    const float bhn = bhh[2 * HDIM + col];
    const int colLoc = c * 16 + nloc;        // [0,64) within block
    const int ktl  = colLoc >> 5;
    const int kloc = colLoc & 31;
    #pragma unroll
    for (int v = 0; v < 8; ++v) {
      const int ro = mbase + v + 8 * khalf;
      float r  = sigmoidf_(r_acc[c][v] + brz);
      float zz = sigmoidf_(z_acc[c][v] + bzz);
      float nn = tanhf(xn_acc[c][v] + bxn + r * (hn_acc[c][v] + bhn));
      float hold = h_in[(size_t)ro * HDIM + col];
      float hnew = (1.0f - zz) * nn + zz * hold;
      h_out[(size_t)ro * HDIM + col] = hnew;
      if constexpr (IS_ENC) {
        if (endp[ro] == t) hf[(size_t)ro * HDIM + col] = hnew;
      }
      int lp, e;
      amap(v + 8 * khalf, kloc, lp, e);
      stile[(wave << 1) + ktl][lp * 16 + e] = f2bf(hnew);
    }
  }
  __syncthreads();

  // coalesced copy-out of the permuted bf16 tiles (16 x 1KB)
  {
    const int chunk = threadIdx.x >> 4;    // tile chunk: (mt<<1)|ktl
    const int part  = threadIdx.x & 15;    // 32 bf16 per part
    const int mtL = chunk >> 1;
    const int ktg = blockIdx.x * 2 + (chunk & 1);
    const uint4* src = (const uint4*)&stile[chunk][part * 32];
    uint4 q0 = src[0], q1 = src[1], q2 = src[2], q3 = src[3];
    uint4* dst = (uint4*)(hP_out + (((size_t)mtL * KT_H + ktg) << 9) + part * 32);
    dst[0] = q0; dst[1] = q1; dst[2] = q2; dst[3] = q3;
    if constexpr (!IS_ENC) {
      uint4* dst2 = (uint4*)(seqP + (((size_t)(t * 8 + mtL) * KT_H + ktg) << 9) + part * 32);
      dst2[0] = q0; dst2[1] = q1; dst2[2] = q2; dst2[3] = q3;
    }
  }
}

// ---------------- generic GEMM: out = act(A[M,K] @ Wp^T + bias) --------------
// A either permuted bf16 (A_PERM) or plain fp32 rows. Optional fp32 output Cf
// and/or permuted-bf16 output Cp (LDS-staged) for chaining GEMMs.
template <bool A_PERM>
__global__ __launch_bounds__(256) void gemm_kernel(
    const void* __restrict__ Ain, const bf16_t* __restrict__ Wp,
    const float* __restrict__ bias, float* __restrict__ Cf,
    bf16_t* __restrict__ Cp, int K, int Ncols, int act) {
  __shared__ bf16_t stile[16][512];
  const int lane  = threadIdx.x & 31;
  const int wave  = threadIdx.x >> 5;
  const int khalf = lane >> 4;
  const int nloc  = lane & 15;
  const int mbase = blockIdx.x * 128 + wave * 16;
  const int colbase = blockIdx.y * 64;
  const int ktiles = K >> 5;

  v8f acc[4];
  #pragma unroll
  for (int c = 0; c < 4; ++c) { v8f z = {0,0,0,0,0,0,0,0}; acc[c] = z; }

  for (int kt = 0; kt < ktiles; ++kt) {
    v16bf af;
    if constexpr (A_PERM) {
      const bf16_t* Ap = (const bf16_t*)Ain;
      af = load_tile(Ap, (size_t)(blockIdx.x * 8 + wave) * ktiles + kt, lane);
    } else {
      const float* A = (const float*)Ain;
      af = load_a_frag_f32(A + (size_t)(mbase + nloc) * K, kt * 32, khalf);
    }
    #pragma unroll
    for (int c = 0; c < 4; ++c) {
      v16bf bf_ = load_tile(Wp, (size_t)((colbase >> 4) + c) * ktiles + kt, lane);
      acc[c] = wmma_bf16(af, bf_, acc[c]);
    }
  }

  #pragma unroll
  for (int c = 0; c < 4; ++c) {
    const int col = colbase + c * 16 + nloc;
    const float b = bias[col];
    const int colLoc = c * 16 + nloc;
    const int ktl  = colLoc >> 5;
    const int kloc = colLoc & 31;
    #pragma unroll
    for (int v = 0; v < 8; ++v) {
      const int ro = mbase + v + 8 * khalf;
      float x = acc[c][v] + b;
      if (act) x = fmaxf(x, 0.0f);
      if (Cf) Cf[(size_t)ro * Ncols + col] = x;
      if (Cp) {
        int lp, e;
        amap(v + 8 * khalf, kloc, lp, e);
        stile[(wave << 1) + ktl][lp * 16 + e] = f2bf(x);
      }
    }
  }
  if (Cp) {
    __syncthreads();
    const int chunk = threadIdx.x >> 4;
    const int part  = threadIdx.x & 15;
    const int mtg = blockIdx.x * 8 + (chunk >> 1);
    const int ktg = blockIdx.y * 2 + (chunk & 1);
    const uint4* src = (const uint4*)&stile[chunk][part * 32];
    uint4* dst = (uint4*)(Cp + (((size_t)mtg * (Ncols >> 5) + ktg) << 9) + part * 32);
    dst[0] = src[0]; dst[1] = src[1]; dst[2] = src[2]; dst[3] = src[3];
  }
}

// ---------------- reparametrize: emb = mu + exp(lv/2)*eps --------------------
// writes fp32 emb (decoder h0) and permuted bf16 (decoder A-side h0)
__global__ void reparam_kernel(const float* __restrict__ mu,
                               const float* __restrict__ lv,
                               const float* __restrict__ eps,
                               float* __restrict__ emb,
                               bf16_t* __restrict__ embP) {
  int i = blockIdx.x * blockDim.x + threadIdx.x;
  if (i >= NBATCH * HDIM) return;
  float v = mu[i] + __expf(0.5f * lv[i]) * eps[i];
  emb[i] = v;
  int row = i >> 10, k = i & 1023;
  int lane, e;
  amap(row & 15, k & 31, lane, e);
  embP[(((size_t)(row >> 4) * KT_H + (k >> 5)) << 9) + lane * 16 + e] = f2bf(v);
}

// ---------------- masked L1 loss, deterministic 2-stage reduction ------------
__global__ __launch_bounds__(256) void loss_partial_kernel(
    const float* __restrict__ yhat,     // [T, N, S]
    const float* __restrict__ s_next,   // [N, T, S]
    const int* __restrict__ startp, float* __restrict__ partial) {
  __shared__ float red[256];
  float sum = 0.0f;
  const int total = T_STEPS * NBATCH * SDIM;
  for (int idx = blockIdx.x * blockDim.x + threadIdx.x; idx < total;
       idx += gridDim.x * blockDim.x) {
    int t   = idx / (NBATCH * SDIM);
    int rem = idx - t * (NBATCH * SDIM);
    int n   = rem >> 7;
    int j   = rem & 127;
    int st  = startp[n];
    if (t < T_STEPS - st) {
      int ts = t + st; if (ts >= T_STEPS) ts -= T_STEPS;
      float ref = s_next[((size_t)n * T_STEPS + ts) * SDIM + j];
      sum += fabsf(ref - yhat[idx]);
    }
  }
  red[threadIdx.x] = sum;
  __syncthreads();
  for (int o = 128; o > 0; o >>= 1) {
    if (threadIdx.x < o) red[threadIdx.x] += red[threadIdx.x + o];
    __syncthreads();
  }
  if (threadIdx.x == 0) partial[blockIdx.x] = red[0];
}

__global__ void loss_final_kernel(const float* __restrict__ partial, int nb,
                                  float* __restrict__ out) {
  __shared__ float red[256];
  float s = 0.0f;
  for (int i = threadIdx.x; i < nb; i += 256) s += partial[i];
  red[threadIdx.x] = s;
  __syncthreads();
  for (int o = 128; o > 0; o >>= 1) {
    if (threadIdx.x < o) red[threadIdx.x] += red[threadIdx.x + o];
    __syncthreads();
  }
  if (threadIdx.x == 0) out[0] = red[0];
}

// ---------------- host-side orchestration ------------------------------------
extern "C" void kernel_launch(void* const* d_in, const int* in_sizes, int n_in,
                              void* d_out, int out_size, void* d_ws,
                              size_t ws_size, hipStream_t stream) {
  (void)in_sizes; (void)n_in; (void)out_size; (void)ws_size;
  const float* s       = (const float*)d_in[0];
  const float* a       = (const float*)d_in[1];
  const float* s_next  = (const float*)d_in[3];
  const float* eps     = (const float*)d_in[4];
  const float* encWih  = (const float*)d_in[5];
  const float* encWhh  = (const float*)d_in[6];
  const float* enc_bih = (const float*)d_in[7];
  const float* enc_bhh = (const float*)d_in[8];
  const float* muW     = (const float*)d_in[9];
  const float* mu_b    = (const float*)d_in[10];
  const float* lvW     = (const float*)d_in[11];
  const float* lv_b    = (const float*)d_in[12];
  const float* decWih  = (const float*)d_in[19];
  const float* decWhh  = (const float*)d_in[20];
  const float* dec_bih = (const float*)d_in[21];
  const float* dec_bhh = (const float*)d_in[22];
  const float* d1W     = (const float*)d_in[23];
  const float* d1_b    = (const float*)d_in[24];
  const float* d2W     = (const float*)d_in[25];
  const float* d2_b    = (const float*)d_in[26];
  const float* d3W     = (const float*)d_in[27];
  const float* d3_b    = (const float*)d_in[28];

  char* w = (char*)d_ws;
  size_t off = 0;
  auto alloc = [&](size_t bytes) -> void* {
    off = (off + 255) & ~(size_t)255;
    void* p = w + off;
    off += bytes;
    return p;
  };
  bf16_t* encWihP = (bf16_t*)alloc((size_t)3072 * 160  * 2);
  bf16_t* encWhhP = (bf16_t*)alloc((size_t)3072 * 1024 * 2);
  bf16_t* decWihP = (bf16_t*)alloc((size_t)3072 * 32   * 2);
  bf16_t* decWhhP = (bf16_t*)alloc((size_t)3072 * 1024 * 2);
  bf16_t* muWP    = (bf16_t*)alloc((size_t)1024 * 1024 * 2);
  bf16_t* lvWP    = (bf16_t*)alloc((size_t)1024 * 1024 * 2);
  bf16_t* d1P     = (bf16_t*)alloc((size_t)128 * 1024 * 2);
  bf16_t* d2P     = (bf16_t*)alloc((size_t)64  * 128  * 2);
  bf16_t* d3P     = (bf16_t*)alloc((size_t)128 * 64   * 2);
  int*    endp    = (int*)alloc(NBATCH * 4);
  int*    startp  = (int*)alloc(NBATCH * 4);
  float*  hA      = (float*)alloc((size_t)NBATCH * HDIM * 4);
  float*  hB      = (float*)alloc((size_t)NBATCH * HDIM * 4);
  bf16_t* hPA     = (bf16_t*)alloc((size_t)NBATCH * HDIM * 2);
  bf16_t* hPB     = (bf16_t*)alloc((size_t)NBATCH * HDIM * 2);
  float*  hf      = (float*)alloc((size_t)NBATCH * HDIM * 4);
  float*  mu      = (float*)alloc((size_t)NBATCH * HDIM * 4);
  float*  lv      = (float*)alloc((size_t)NBATCH * HDIM * 4);
  bf16_t* xPenc   = (bf16_t*)alloc((size_t)T_STEPS * NBATCH * 160 * 2);
  bf16_t* aPdec   = (bf16_t*)alloc((size_t)T_STEPS * NBATCH * ADIM * 2);
  bf16_t* seqP    = (bf16_t*)alloc((size_t)T_STEPS * NBATCH * HDIM * 2);
  bf16_t* y1P     = (bf16_t*)alloc((size_t)T_STEPS * NBATCH * 128 * 2);
  bf16_t* y2P     = (bf16_t*)alloc((size_t)T_STEPS * NBATCH * 64  * 2);
  float*  yhat    = (float*)alloc((size_t)T_STEPS * NBATCH * SDIM * 4);
  float*  partial = (float*)alloc(1024 * 4);

  split_points_kernel<<<1, 128, 0, stream>>>(s, endp, startp);

  auto perm = [&](const float* src, bf16_t* dst, int R, int K) {
    int total = R * K;
    permute_weights_kernel<<<(total + 255) / 256, 256, 0, stream>>>(src, dst, R, K);
  };
  perm(encWih, encWihP, 3072, 160);
  perm(encWhh, encWhhP, 3072, 1024);
  perm(decWih, decWihP, 3072, 32);
  perm(decWhh, decWhhP, 3072, 1024);
  perm(muW,    muWP,    1024, 1024);
  perm(lvW,    lvWP,    1024, 1024);
  perm(d1W,    d1P,     128, 1024);
  perm(d2W,    d2P,     64,  128);
  perm(d3W,    d3P,     128, 64);

  {
    int tot = T_STEPS * NBATCH * 160;
    pack_xenc_kernel<<<(tot + 255) / 256, 256, 0, stream>>>(s, a, xPenc);
  }
  {
    int tot = T_STEPS * NBATCH * ADIM;
    pack_adec_kernel<<<(tot + 255) / 256, 256, 0, stream>>>(a, startp, aPdec);
  }

  // ---- encoder: h0 = ones ----
  init_h_ones_kernel<<<(NBATCH * HDIM + 255) / 256, 256, 0, stream>>>(hA, hPA);
  for (int t = 0; t < T_STEPS; ++t) {
    const float*  hin  = (t & 1) ? hB : hA;
    float*        hout = (t & 1) ? hA : hB;
    const bf16_t* hpin = (t & 1) ? hPB : hPA;
    bf16_t*       hpout= (t & 1) ? hPA : hPB;
    gru_step_kernel<5, true><<<HDIM / 64, 256, 0, stream>>>(
        xPenc, endp, t, hin, hout, hpin, hpout, encWihP, encWhhP,
        enc_bih, enc_bhh, hf, nullptr);
  }

  // ---- latent heads + reparametrization (emb -> decoder h0) ----
  gemm_kernel<false><<<dim3(1, HDIM / 64), 256, 0, stream>>>(
      hf, muWP, mu_b, mu, nullptr, HDIM, HDIM, 0);
  gemm_kernel<false><<<dim3(1, HDIM / 64), 256, 0, stream>>>(
      hf, lvWP, lv_b, lv, nullptr, HDIM, HDIM, 0);
  reparam_kernel<<<(NBATCH * HDIM + 255) / 256, 256, 0, stream>>>(mu, lv, eps,
                                                                  hA, hPA);

  // ---- decoder (emits dec_out in permuted bf16 directly) ----
  for (int t = 0; t < T_STEPS; ++t) {
    const float*  hin  = (t & 1) ? hB : hA;
    float*        hout = (t & 1) ? hA : hB;
    const bf16_t* hpin = (t & 1) ? hPB : hPA;
    bf16_t*       hpout= (t & 1) ? hPA : hPB;
    gru_step_kernel<1, false><<<HDIM / 64, 256, 0, stream>>>(
        aPdec, endp, t, hin, hout, hpin, hpout, decWihP, decWhhP,
        dec_bih, dec_bhh, nullptr, seqP);
  }

  // ---- decoder MLP head chain, bf16-permuted between stages ----
  const int M = T_STEPS * NBATCH;
  gemm_kernel<true><<<dim3(M / 128, 128 / 64), 256, 0, stream>>>(
      seqP, d1P, d1_b, nullptr, y1P, 1024, 128, 1);
  gemm_kernel<true><<<dim3(M / 128, 64 / 64), 256, 0, stream>>>(
      y1P, d2P, d2_b, nullptr, y2P, 128, 64, 1);
  gemm_kernel<true><<<dim3(M / 128, 128 / 64), 256, 0, stream>>>(
      y2P, d3P, d3_b, yhat, nullptr, 64, 128, 0);

  // ---- masked L1 loss ----
  loss_partial_kernel<<<1024, 256, 0, stream>>>(yhat, s_next, startp, partial);
  loss_final_kernel<<<1, 256, 0, stream>>>(partial, 1024, (float*)d_out);
}
